// LSTMClassifier_9904194585008
// MI455X (gfx1250) — matrix operand order repaired and checked by reading, not compiled, not run on hardware
//
#include <hip/hip_runtime.h>
#include <hip/hip_bf16.h>

// Problem sizes (fixed by the reference).
#define Vv   24
#define Hh   512
#define Tt   512
#define Bb   256
#define Cc   10
#define G4H  2048   // 4*H

typedef __bf16 v16bf __attribute__((ext_vector_type(16)));
typedef __bf16 v8bf  __attribute__((ext_vector_type(8)));
typedef float  v8f   __attribute__((ext_vector_type(8)));

__device__ __forceinline__ unsigned short f32_to_bf16_rne(float x) {
    union { float f; unsigned int u; } v; v.f = x;
    unsigned int u = v.u;
    u += 0x7FFFu + ((u >> 16) & 1u);   // round-to-nearest-even
    return (unsigned short)(u >> 16);
}
__device__ __forceinline__ float sigm_f(float x)  { return 1.0f / (1.0f + __expf(-x)); }
__device__ __forceinline__ float tanh_f(float x)  { return 2.0f / (1.0f + __expf(-2.0f * x)) - 1.0f; }

// ---------------------------------------------------------------------------
// Kernel 1: W_hh (f32 [2048][512] row-major) -> bf16 row-major. 2 MB, L2-resident.
// ---------------------------------------------------------------------------
__global__ void pack_whh_bf16(const float* __restrict__ Whh, unsigned short* __restrict__ Wbf) {
    int i = blockIdx.x * blockDim.x + threadIdx.x;
    if (i < G4H * Hh) Wbf[i] = f32_to_bf16_rne(Whh[i]);
}

// ---------------------------------------------------------------------------
// Kernel 2: fused one-hot projection table proj[v][g] = W_ih[g][v] + b_ih[g] + b_hh[g].
// ---------------------------------------------------------------------------
__global__ void build_proj(const float* __restrict__ Wih, const float* __restrict__ bih,
                           const float* __restrict__ bhh, float* __restrict__ proj) {
    int i = blockIdx.x * blockDim.x + threadIdx.x;      // i = v*G4H + g
    if (i < Vv * G4H) {
        int v = i / G4H, g = i % G4H;
        proj[i] = Wih[g * Vv + v] + bih[g] + bhh[g];
    }
}

// ---------------------------------------------------------------------------
// Kernel 3: persistent LSTM. grid = B/16 workgroups, 512 threads (16 wave32s).
// Each WG owns a 16-row batch tile; h state lives in LDS (bf16, ping-pong),
// c state lives in registers. Per step: gates = proj_gather + h @ Whh^T via
// v_wmma_f32_16x16x32_bf16, then the elementwise cell update, two barriers.
// Wave w owns hidden columns [w*32, w*32+32): 4 gates x 2 N-tiles of 16.
// ---------------------------------------------------------------------------
__global__ __launch_bounds__(512, 1)
void lstm_persist(const int* __restrict__ X, const int* __restrict__ lengths,
                  const unsigned short* __restrict__ Wbf, const float* __restrict__ proj,
                  float* __restrict__ h_last)
{
    __shared__ unsigned short hbuf[2][16][Hh];   // 32 KB ping-pong h state (bf16)
    __shared__ int tok_sh[16];

    const int tid    = threadIdx.x;
    const int lane   = tid & 31;
    const int wave   = tid >> 5;
    const int nloc   = lane & 15;        // N (and A-row M) position within 16
    const int hiHalf = lane >> 4;        // lane group 0..15 vs 16..31
    const int jw     = wave * 32;        // this wave's hidden-column base
    const int b0     = blockIdx.x * 16;  // batch tile base

    // zero the t=0 h buffer (h0 = 0)
    for (int i = tid; i < 16 * Hh; i += 512) (&hbuf[0][0][0])[i] = 0;

    // Accumulator row mapping (C/D layout): vgpr r -> M = r (lanes 0-15) / r+8 (lanes 16-31)
    int mrow[8]; int lenm[8];
    #pragma unroll
    for (int r = 0; r < 8; ++r) {
        mrow[r] = hiHalf ? (r + 8) : r;
        lenm[r] = lengths[b0 + mrow[r]];
    }

    // cell state in registers, same element mapping as accumulators
    v8f cst[2];
    #pragma unroll
    for (int jt = 0; jt < 2; ++jt)
        #pragma unroll
        for (int r = 0; r < 8; ++r) cst[jt][r] = 0.0f;

    for (int t = 0; t < Tt; ++t) {
        // stage this step's 16 token ids
        if (tid < 16) tok_sh[tid] = X[(b0 + tid) * Tt + t];
        __syncthreads();   // h writes of prev step + tokens visible

        const unsigned short* __restrict__ hc = &hbuf[t & 1][0][0];
        unsigned short* __restrict__ hn       = &hbuf[(t + 1) & 1][0][0];

        int tokm[8];
        #pragma unroll
        for (int r = 0; r < 8; ++r) tokm[r] = tok_sh[mrow[r]];

        // init accumulators with the fused one-hot projection (gather from L2-resident table)
        v8f acc[4][2];
        #pragma unroll
        for (int G = 0; G < 4; ++G)
            #pragma unroll
            for (int jt = 0; jt < 2; ++jt) {
                const int g = G * Hh + jw + jt * 16 + nloc;
                #pragma unroll
                for (int r = 0; r < 8; ++r)
                    acc[G][jt][r] = proj[tokm[r] * G4H + g];
            }

        // GEMM: K = 512 in 16 slices of 32; A fragment (h rows) reused across 8 tiles
        #pragma unroll 4
        for (int kt = 0; kt < 16; ++kt) {
            // A fragment, contiguous K order (matches B):
            // lanes 0-15 : halves 0-15 -> K 0..15
            // lanes 16-31: halves 0-15 -> K 16..31
            const int abase = nloc * Hh + kt * 32;
            const v8bf a0 = *reinterpret_cast<const v8bf*>(&hc[abase + (hiHalf ? 16 : 0)]);
            const v8bf a1 = *reinterpret_cast<const v8bf*>(&hc[abase + (hiHalf ? 24 : 8)]);
            const v16bf a = __builtin_shufflevector(a0, a1,
                0, 1, 2, 3, 4, 5, 6, 7, 8, 9, 10, 11, 12, 13, 14, 15);

            #pragma unroll
            for (int G = 0; G < 4; ++G)
                #pragma unroll
                for (int jt = 0; jt < 2; ++jt) {
                    const int g = G * Hh + jw + jt * 16 + nloc;  // this lane's B column
                    // B fragment: lane holds 16 contiguous K values of column g
                    const v16bf bf = *reinterpret_cast<const v16bf*>(
                        &Wbf[g * Hh + kt * 32 + hiHalf * 16]);
                    acc[G][jt] = __builtin_amdgcn_wmma_f32_16x16x32_bf16(
                        false, a, false, bf, (short)0, acc[G][jt], false, false);
                }
        }

        // cell update (torch gate order i,f,g,o) entirely in registers
        #pragma unroll
        for (int jt = 0; jt < 2; ++jt) {
            const int j = jw + jt * 16 + nloc;
            #pragma unroll
            for (int r = 0; r < 8; ++r) {
                const float ig = sigm_f(acc[0][jt][r]);
                const float fg = sigm_f(acc[1][jt][r]);
                const float gg = tanh_f(acc[2][jt][r]);
                const float og = sigm_f(acc[3][jt][r]);
                const float cv = fg * cst[jt][r] + ig * gg;
                cst[jt][r] = cv;
                const float hv = og * tanh_f(cv);
                hn[mrow[r] * Hh + j] = f32_to_bf16_rne(hv);
                if (t == lenm[r] - 1)                         // packed-sequence last state
                    h_last[(b0 + mrow[r]) * Hh + j] = hv;
            }
        }
        __syncthreads();   // h(t+1) complete before anyone reads it / restages tokens
    }
}

// ---------------------------------------------------------------------------
// Kernel 4: final classifier, out[b][c] = h_last[b] . fc_W[c] + fc_b[c]  (2.7 MFLOP)
// ---------------------------------------------------------------------------
__global__ void fc_kernel(const float* __restrict__ h_last, const float* __restrict__ fcW,
                          const float* __restrict__ fcb, float* __restrict__ out) {
    int i = blockIdx.x * blockDim.x + threadIdx.x;
    if (i >= Bb * Cc) return;
    int b = i / Cc, c = i % Cc;
    const float* h = h_last + b * Hh;
    const float* w = fcW + c * Hh;
    float s = fcb[c];
    #pragma unroll 8
    for (int k = 0; k < Hh; ++k) s += h[k] * w[k];
    out[i] = s;
}

extern "C" void kernel_launch(void* const* d_in, const int* in_sizes, int n_in,
                              void* d_out, int out_size, void* d_ws, size_t ws_size,
                              hipStream_t stream) {
    const int*   X       = (const int*)d_in[0];
    const int*   lengths = (const int*)d_in[1];
    const float* W_ih    = (const float*)d_in[2];
    const float* W_hh    = (const float*)d_in[3];
    const float* b_ih    = (const float*)d_in[4];
    const float* b_hh    = (const float*)d_in[5];
    const float* fc_W    = (const float*)d_in[6];
    const float* fc_b    = (const float*)d_in[7];
    float* out = (float*)d_out;

    char* ws = (char*)d_ws;
    unsigned short* Wbf  = (unsigned short*)ws;                              // 2 MB
    float* proj          = (float*)(ws + (size_t)G4H * Hh * 2);              // 192 KB
    float* h_last        = (float*)(ws + (size_t)G4H * Hh * 2
                                       + (size_t)Vv * G4H * 4);              // 512 KB

    pack_whh_bf16<<<(G4H * Hh + 255) / 256, 256, 0, stream>>>(W_hh, Wbf);
    build_proj  <<<(Vv * G4H + 255) / 256, 256, 0, stream>>>(W_ih, b_ih, b_hh, proj);
    lstm_persist<<<Bb / 16, 512, 0, stream>>>(X, lengths, Wbf, proj, h_last);
    fc_kernel   <<<(Bb * Cc + 255) / 256, 256, 0, stream>>>(h_last, fc_W, fc_b, out);
}
